// AttBlockMinimal_33509334843499
// MI455X (gfx1250) — matrix operand order repaired and checked by reading, not compiled, not run on hardware
//
#include <hip/hip_runtime.h>

// ---------------- types ----------------
typedef __attribute__((ext_vector_type(16))) __bf16          v16bf;
typedef __attribute__((ext_vector_type(8)))  float           v8f;
typedef __attribute__((ext_vector_type(4)))  float           f32x4;
typedef __attribute__((ext_vector_type(8)))  unsigned short  u16x8;
typedef __attribute__((ext_vector_type(16))) unsigned short  u16x16;
typedef __attribute__((ext_vector_type(8)))  unsigned int    u32x8;
typedef __attribute__((ext_vector_type(4)))  unsigned int    u32x4;
typedef __attribute__((__vector_size__(16))) int             vi4;   // async builtin param type

#define AS1 __attribute__((address_space(1)))
#define AS3 __attribute__((address_space(3)))

#define NXc   8192
#define NFc   8192
#define DQK   128
#define DV    256
#define NEGINF (-1.0e9f)

// ---- bf16 helpers: round-by-bias + v_perm pack (2 elems / instr) ----
__device__ __forceinline__ unsigned pack2bf(float lo, float hi) {
  unsigned a = __builtin_bit_cast(unsigned, lo) + 0x8000u;
  unsigned b = __builtin_bit_cast(unsigned, hi) + 0x8000u;
  return __builtin_amdgcn_perm(b, a, 0x07060302u); // {b[31:16], a[31:16]}
}
__device__ __forceinline__ unsigned short bf1(float f) {
  unsigned u = __builtin_bit_cast(unsigned, f) + 0x8000u;
  return (unsigned short)(u >> 16);
}

// ---- async global->LDS 16B copy (CDNA5 ASYNCcnt path, guarded) ----
__device__ __forceinline__ void copy16B(const unsigned short* g, unsigned short* l) {
#if __has_builtin(__builtin_amdgcn_global_load_async_to_lds_b128)
  const AS1 void* gc = (const AS1 void*)g;
  AS3 void*       lc = (AS3 void*)l;
  __builtin_amdgcn_global_load_async_to_lds_b128((AS1 vi4*)gc, (AS3 vi4*)lc, 0, 0);
#else
  *(u16x8*)l = *(const u16x8*)g;
#endif
}
__device__ __forceinline__ void wait_async() {
#if __has_builtin(__builtin_amdgcn_s_wait_asynccnt)
  __builtin_amdgcn_s_wait_asynccnt(0);
#endif
}

// ============================================================
// Projection GEMM: C_bf16[M,N] (or C^T if TRANS) = A_f32[M,K] @ W_f32[K,N]
// one wave per 16x16 tile, v_wmma_f32_16x16x32_bf16
// ============================================================
template <bool TRANS>
__global__ __launch_bounds__(128) void proj_gemm_bf16(
    const float* __restrict__ A, const float* __restrict__ W,
    unsigned short* __restrict__ C, int M, int K, int N) {
  const int lane = threadIdx.x & 31;
  const int wid  = blockIdx.x * (blockDim.x >> 5) + (threadIdx.x >> 5);
  const int ntn  = N >> 4;
  const int tm   = wid / ntn;
  const int tn   = wid % ntn;
  if (tm >= (M >> 4)) return;

  const bool hi  = lane >= 16;
  const int  r16 = lane & 15;

  v8f acc = {};
  const float* arow = A + (size_t)(tm * 16 + r16) * K;
  const int colW = tn * 16 + r16;

  for (int kc = 0; kc < K; kc += 32) {
    // A fragment: lane<16: K=kc+{0..7,16..23}; lane>=16: K=kc+{8..15,24..31}
    const int ab = kc + (hi ? 8 : 0);
    f32x4 a0 = *(const f32x4*)(arow + ab);
    f32x4 a1 = *(const f32x4*)(arow + ab + 4);
    f32x4 a2 = *(const f32x4*)(arow + ab + 16);
    f32x4 a3 = *(const f32x4*)(arow + ab + 20);
    u32x8 ap;
    ap[0] = pack2bf(a0[0], a0[1]); ap[1] = pack2bf(a0[2], a0[3]);
    ap[2] = pack2bf(a1[0], a1[1]); ap[3] = pack2bf(a1[2], a1[3]);
    ap[4] = pack2bf(a2[0], a2[1]); ap[5] = pack2bf(a2[2], a2[3]);
    ap[6] = pack2bf(a3[0], a3[1]); ap[7] = pack2bf(a3[2], a3[3]);
    v16bf a = __builtin_bit_cast(v16bf, ap);
    // B fragment: lane<16: K=kc+{0..7,16..23} ; lane>=16: K=kc+{8..15,24..31} ; col = lane&15
    const int kb = kc + (hi ? 8 : 0);
    float wv[16];
#pragma unroll
    for (int j = 0; j < 16; ++j) wv[j] = W[(size_t)(kb + j + (j >= 8 ? 8 : 0)) * N + colW];
    u32x8 bp;
#pragma unroll
    for (int j = 0; j < 8; ++j) bp[j] = pack2bf(wv[2 * j], wv[2 * j + 1]);
    v16bf b = __builtin_bit_cast(v16bf, bp);
    acc = __builtin_amdgcn_wmma_f32_16x16x32_bf16(false, a, false, b,
                                                  (short)0, acc, false, false);
  }

  if (TRANS) {
    // C^T[n][m]: along r the rows m are consecutive -> one b128 store
    u32x4 st;
#pragma unroll
    for (int i = 0; i < 4; ++i) st[i] = pack2bf(acc[2 * i], acc[2 * i + 1]);
    unsigned short* p = C + (size_t)colW * M + tm * 16 + (hi ? 8 : 0);
    *(u32x4*)p = st;
  } else {
#pragma unroll
    for (int r = 0; r < 8; ++r) {
      const int m = tm * 16 + r + (hi ? 8 : 0);
      C[(size_t)m * N + colW] = bf1(acc[r]);
    }
  }
}

// ============================================================
// Fused masked attention, 4 waves/block, one 16-row strip per wave.
// K tiles (32 keys) and V slabs staged once per block via async-to-LDS.
// ============================================================
__global__ __launch_bounds__(128, 1) void attn_kernel(
    const unsigned short* __restrict__ Qb,   // [8192][128] bf16
    const unsigned short* __restrict__ Kb,   // [8192][128] bf16
    const unsigned short* __restrict__ Vt,   // [256][8192] bf16 (V^T)
    const int* __restrict__ x_batch, const int* __restrict__ f_batch,
    float* ctx, float* attn) {
  __shared__ __align__(16) unsigned short Ksh[32 * 136];   // 32-key K tile, padded
  __shared__ __align__(16) unsigned short Vsh[256 * 40];   // V slab (32 keys)
  __shared__ __align__(16) float pshare[4][16 * 36];       // P tiles per wave
  __shared__ float msh[4][16];
  __shared__ float invsh[4][16];

  const int lane = threadIdx.x & 31;
  const int w    = threadIdx.x >> 5;
  const int wid  = blockIdx.x * 4 + w;   // 512 strips
  const int q0   = wid * 16;
  const bool hi  = lane >= 16;
  const int  r16 = lane & 15;

  int xb[8];
#pragma unroll
  for (int r = 0; r < 8; ++r) xb[r] = x_batch[q0 + r + (hi ? 8 : 0)];

  // Q A-fragments (d=128 -> 4 chunks of K=32), raw b128 loads
  v16bf qa[4];
  const unsigned short* qrow = Qb + (size_t)(q0 + r16) * DQK;
#pragma unroll
  for (int c = 0; c < 4; ++c) {
    const int ab = c * 32 + (hi ? 16 : 0);
    u16x8 lo8 = *(const u16x8*)(qrow + ab);
    u16x8 hi8 = *(const u16x8*)(qrow + ab + 8);
    u16x16 cb;
#pragma unroll
    for (int j = 0; j < 8; ++j) { cb[j] = lo8[j]; cb[8 + j] = hi8[j]; }
    qa[c] = __builtin_bit_cast(v16bf, cb);
  }

  float mrow[8];                         // per-LANE running max (no shuffles in loop)
#pragma unroll
  for (int r = 0; r < 8; ++r) mrow[r] = -3.0e38f;

  // ---------------- Phase 1: masked scores + per-lane running max ----------------
  for (int kt2 = 0; kt2 < NFc / 32; ++kt2) {
    const int col0 = kt2 * 32;
    __syncthreads();                      // Ksh reuse guard
    {                                     // stage 32 K rows: 512 x 16B chunks
      const int t = threadIdx.x;
#pragma unroll
      for (int i = 0; i < 4; ++i) {
        const int c   = t + i * 128;
        const int row = c >> 4;
        const int c8  = c & 15;
        copy16B(Kb + (size_t)(col0 + row) * DQK + c8 * 8,
                &Ksh[row * 136 + c8 * 8]);
      }
    }
    wait_async();
    __syncthreads();

#pragma unroll
    for (int half = 0; half < 2; ++half) {
      const int colh = col0 + half * 16;
      v8f s = {};
#pragma unroll
      for (int c = 0; c < 4; ++c) {
        const int kb = c * 32 + (hi ? 16 : 0);
        u16x16 braw = *(const u16x16*)&Ksh[(half * 16 + r16) * 136 + kb];
        s = __builtin_amdgcn_wmma_f32_16x16x32_bf16(
            false, qa[c], false, __builtin_bit_cast(v16bf, braw),
            (short)0, s, false, false);
      }
      const int fb = f_batch[colh + r16];
#pragma unroll
      for (int r = 0; r < 8; ++r) {
        float v = (xb[r] != fb) ? NEGINF : s[r];
        mrow[r] = fmaxf(mrow[r], v);      // per-lane only; reduce after loop
        attn[(size_t)(q0 + r + (hi ? 8 : 0)) * NFc + colh + r16] = v;
      }
    }
  }
  // one cross-lane reduction per row (was per-tile before)
#pragma unroll
  for (int r = 0; r < 8; ++r) {
    float tv = mrow[r];
    tv = fmaxf(tv, __shfl_xor(tv, 1, 32));
    tv = fmaxf(tv, __shfl_xor(tv, 2, 32));
    tv = fmaxf(tv, __shfl_xor(tv, 4, 32));
    tv = fmaxf(tv, __shfl_xor(tv, 8, 32));
    mrow[r] = tv;
  }

  // bridge row max C-layout -> row-per-lane layout via LDS
  if (r16 == 0) {
#pragma unroll
    for (int r = 0; r < 8; ++r) msh[w][(hi ? 8 : 0) + r] = mrow[r];
  }
  const int prow = lane & 15;   // strip row this lane owns in phases 2/3
  const int pseg = lane >> 4;   // column half (2 lanes per row)
  const float mr = msh[w][prow];

  // ---------------- Phase 2: exp + P@V ----------------
  v8f acc[16];
  {
    v8f z = {};
#pragma unroll
    for (int v = 0; v < 16; ++v) acc[v] = z;
  }
  float lsum = 0.0f;
  float* mysh = pshare[w];

  for (int kp = 0; kp < NFc / 32; ++kp) {
    const int key0 = kp * 32;
    __syncthreads();                      // Vsh reuse guard
    {                                     // stage V slab: 1024 x 16B chunks
      const int t = threadIdx.x;
#pragma unroll
      for (int i = 0; i < 8; ++i) {
        const int c   = t + i * 128;
        const int col = c >> 2;
        const int kg  = c & 3;
        copy16B(Vt + (size_t)col * NFc + key0 + kg * 8,
                &Vsh[col * 40 + kg * 8]);
      }
    }
    // overlap the async V staging with the exp pass over S (L2-hot)
    {
      float* base = attn + (size_t)(q0 + prow) * NFc + key0 + pseg * 16;
#pragma unroll
      for (int j4 = 0; j4 < 4; ++j4) {
        f32x4 sv = *(const f32x4*)(base + j4 * 4);
        f32x4 p;
#pragma unroll
        for (int e = 0; e < 4; ++e) p[e] = __expf(sv[e] - mr);
        lsum += p[0] + p[1] + p[2] + p[3];
        *(f32x4*)(base + j4 * 4) = p;
        *(f32x4*)&mysh[prow * 36 + pseg * 16 + j4 * 4] = p;
      }
    }
    wait_async();
    __syncthreads();

    // P A-fragment from LDS (row = lane&15), pack to bf16
    v16bf pa;
    {
      const int ab = hi ? 16 : 0;
      f32x4 p0 = *(const f32x4*)&mysh[r16 * 36 + ab];
      f32x4 p1 = *(const f32x4*)&mysh[r16 * 36 + ab + 4];
      f32x4 p2 = *(const f32x4*)&mysh[r16 * 36 + ab + 8];
      f32x4 p3 = *(const f32x4*)&mysh[r16 * 36 + ab + 12];
      u32x8 pk;
      pk[0] = pack2bf(p0[0], p0[1]); pk[1] = pack2bf(p0[2], p0[3]);
      pk[2] = pack2bf(p1[0], p1[1]); pk[3] = pack2bf(p1[2], p1[3]);
      pk[4] = pack2bf(p2[0], p2[1]); pk[5] = pack2bf(p2[2], p2[3]);
      pk[6] = pack2bf(p3[0], p3[1]); pk[7] = pack2bf(p3[2], p3[3]);
      pa = __builtin_bit_cast(v16bf, pk);
    }
    // context += P @ V, B-fragments straight out of LDS
#pragma unroll
    for (int v = 0; v < 16; ++v) {
      const int colv = v * 16 + r16;
      const int kb = hi ? 16 : 0;
      u16x16 braw = *(const u16x16*)&Vsh[colv * 40 + kb];
      acc[v] = __builtin_amdgcn_wmma_f32_16x16x32_bf16(
          false, pa, false, __builtin_bit_cast(v16bf, braw),
          (short)0, acc[v], false, false);
    }
  }

  // row sums: combine the two column-halves, then 1/l
  float tsum = lsum + __shfl_xor(lsum, 16, 32);
  const float invr = 1.0f / tsum;     // fully-masked rows: p==1 -> uniform
  if (pseg == 0) invsh[w][prow] = invr;

  // ---------------- Phase 3: normalize attn strip (b128) + context ----------------
  {
    float* rowp = attn + (size_t)(q0 + prow) * NFc;
    for (int i = 0; i < NFc / 8; ++i) {
      const int off = (i * 2 + pseg) * 4;
      f32x4 v4 = *(const f32x4*)(rowp + off);
#pragma unroll
      for (int e = 0; e < 4; ++e) v4[e] *= invr;
      *(f32x4*)(rowp + off) = v4;
    }
  }
  float invC[8];
#pragma unroll
  for (int r = 0; r < 8; ++r) invC[r] = invsh[w][r + (hi ? 8 : 0)];
#pragma unroll
  for (int v = 0; v < 16; ++v) {
    const int coln = v * 16 + r16;
#pragma unroll
    for (int r = 0; r < 8; ++r) {
      const int m = q0 + r + (hi ? 8 : 0);
      ctx[(size_t)m * DV + coln] = acc[v][r] * invC[r];
    }
  }
}

// ============================================================
// host launcher
// ============================================================
extern "C" void kernel_launch(void* const* d_in, const int* in_sizes, int n_in,
                              void* d_out, int out_size, void* d_ws, size_t ws_size,
                              hipStream_t stream) {
  (void)in_sizes; (void)n_in; (void)out_size; (void)ws_size;
  // order: f, x, h, edge_index, edge_type, f_batch, x_batch, x_init, W_h, W_key, W_value
  const float* f       = (const float*)d_in[0];
  const float* h       = (const float*)d_in[2];
  const int*   f_batch = (const int*)  d_in[5];
  const int*   x_batch = (const int*)  d_in[6];
  const float* W_h     = (const float*)d_in[8];
  const float* W_key   = (const float*)d_in[9];
  const float* W_value = (const float*)d_in[10];

  unsigned short* Qb = (unsigned short*)d_ws;          // [8192][128]
  unsigned short* Kb = Qb + (size_t)NXc * DQK;         // [8192][128]
  unsigned short* Vt = Kb + (size_t)NFc * DQK;         // [256][8192] (V^T)

  float* ctx  = (float*)d_out;                  // [8192][256]
  float* attn = ctx + (size_t)NXc * DV;         // [8192][8192]

  proj_gemm_bf16<false><<<1024, 128, 0, stream>>>(h, W_h,     Qb, NXc, DV,  DQK);
  proj_gemm_bf16<false><<<1024, 128, 0, stream>>>(f, W_key,   Kb, NFc, 512, DQK);
  proj_gemm_bf16<true ><<<2048, 128, 0, stream>>>(f, W_value, Vt, NFc, 512, DV);
  attn_kernel<<<128, 128, 0, stream>>>(Qb, Kb, Vt, x_batch, f_batch, ctx, attn);
}
